// RNNCell_47356309405867
// MI455X (gfx1250) — compile-verified, matrix operand order checked by reading, not run on hardware
//
#include <hip/hip_runtime.h>
#include <math.h>

typedef __attribute__((ext_vector_type(16))) __bf16 v16bf;
typedef __attribute__((ext_vector_type(8)))  __bf16 v8bf;
typedef __attribute__((ext_vector_type(8)))  float  v8f;
typedef __attribute__((ext_vector_type(4)))  float  v4f;

static constexpr int SEQ   = 512;
static constexpr int BATCH = 128;
static constexpr int HID   = 1024;

__device__ __forceinline__ float fast_tanh(float x) {
#if __has_builtin(__builtin_amdgcn_tanhf)
    return __builtin_amdgcn_tanhf(x);   // single v_tanh_f32 (TRANS32)
#else
    return tanhf(x);
#endif
}

// One-time (per launch) weight conversion: W f32 [HID,HID] -> bf16 in d_ws.
__global__ __launch_bounds__(256) void convert_w_kernel(
    const float* __restrict__ W, __bf16* __restrict__ Wb)
{
    const size_t base = ((size_t)blockIdx.x * blockDim.x + threadIdx.x) * 8;
    v4f lo = *(const v4f*)(W + base);
    v4f hi = *(const v4f*)(W + base + 4);
    v8bf o;
    #pragma unroll
    for (int i = 0; i < 4; ++i) {
        o[i]     = (__bf16)lo[i];
        o[4 + i] = (__bf16)hi[i];
    }
    *(v8bf*)(Wb + base) = o;
}

// ---------------------------------------------------------------------------
// One timestep: h_out[b][j] = tanh( x_t[b][j] + sum_k h_prev[b][k]*W[j][k] + b[j] )
// Grid: 128 blocks x 64 threads (2 waves). Each wave: 1 M-tile (16 batch rows)
// x 2 N-tiles (32 hidden cols). K=1024 swept by 32-wide bf16 WMMA.
// PRECONV=true: W already bf16 (from d_ws). PRECONV=false: convert on the fly.
// ---------------------------------------------------------------------------
template <bool PRECONV>
__global__ __launch_bounds__(64) void rnn_step_kernel(
    const float* __restrict__ x_t,      // [BATCH, HID]
    const float* __restrict__ h_prev,   // [BATCH, HID] f32
    const void*  __restrict__ Wp,       // [HID, HID] bf16 (PRECONV) or f32
    const float* __restrict__ bias,     // [HID]
    float* __restrict__ h_out)          // [BATCH, HID]
{
    const int lane = threadIdx.x & 31;
    const int wid  = (blockIdx.x * blockDim.x + threadIdx.x) >> 5; // 0..255

    const int m_tile = wid >> 5;        // 0..7   (BATCH/16)
    const int n_grp  = wid & 31;        // 0..31  (HID/32)
    const int n0     = n_grp * 32;

    const int lr  = lane & 15;          // row-in-tile (A) / col-in-tile (B,C)
    const int sel = lane >> 4;          // K-half selector (16-bit WMMA layout)

    const int col0 = n0 + lr;
    const int col1 = n0 + 16 + lr;

    // Prefetch the epilogue's x_t tile now: input_proj streams from HBM
    // (never L2-resident), and its latency would otherwise sit at the end of
    // every step on the serial scan chain. global_prefetch_b8 has no counter,
    // so this fully overlaps with the K=1024 GEMM below.
    #pragma unroll
    for (int r = 0; r < 8; ++r) {
        const int row = m_tile * 16 + r + sel * 8;
        __builtin_prefetch(x_t + (size_t)row * HID + col0, 0, 3);
    }
    __builtin_prefetch(bias + col0, 0, 3);

    const float* arow = h_prev + (size_t)(m_tile * 16 + lr) * HID;

    v8f c0 = {};
    v8f c1 = {};

    if (PRECONV) {
        const __bf16* Wb = (const __bf16*)Wp;
        const __bf16* brow0 = Wb + (size_t)(n0      + lr) * HID;
        const __bf16* brow1 = Wb + (size_t)(n0 + 16 + lr) * HID;

        #pragma unroll 2
        for (int k0 = 0; k0 < HID; k0 += 32) {
            const int kb = k0 + sel * 8;

            // A fragment: f32 h rows -> bf16. halves 0-7 = K kb..kb+7,
            // halves 8-15 = K kb+16..kb+23.
            v4f a0 = *(const v4f*)(arow + kb);
            v4f a1 = *(const v4f*)(arow + kb + 4);
            v4f a2 = *(const v4f*)(arow + kb + 16);
            v4f a3 = *(const v4f*)(arow + kb + 20);
            v16bf a;
            #pragma unroll
            for (int i = 0; i < 4; ++i) {
                a[i]      = (__bf16)a0[i];
                a[4 + i]  = (__bf16)a1[i];
                a[8 + i]  = (__bf16)a2[i];
                a[12 + i] = (__bf16)a3[i];
            }

            // B fragments: column n of B = row n of W (C = h*W^T); pure loads.
            v8bf p00 = *(const v8bf*)(brow0 + kb);
            v8bf p01 = *(const v8bf*)(brow0 + kb + 16);
            v8bf p10 = *(const v8bf*)(brow1 + kb);
            v8bf p11 = *(const v8bf*)(brow1 + kb + 16);
            v16bf bv0 = __builtin_shufflevector(p00, p01,
                0,1,2,3,4,5,6,7,8,9,10,11,12,13,14,15);
            v16bf bv1 = __builtin_shufflevector(p10, p11,
                0,1,2,3,4,5,6,7,8,9,10,11,12,13,14,15);

            c0 = __builtin_amdgcn_wmma_f32_16x16x32_bf16(
                     false, a, false, bv0, (short)0, c0, false, false);
            c1 = __builtin_amdgcn_wmma_f32_16x16x32_bf16(
                     false, a, false, bv1, (short)0, c1, false, false);
        }
    } else {
        const float* Wf = (const float*)Wp;
        const float* brow0 = Wf + (size_t)(n0      + lr) * HID;
        const float* brow1 = Wf + (size_t)(n0 + 16 + lr) * HID;

        #pragma unroll 2
        for (int k0 = 0; k0 < HID; k0 += 32) {
            const int kb = k0 + sel * 8;
            v4f a0 = *(const v4f*)(arow + kb);
            v4f a1 = *(const v4f*)(arow + kb + 4);
            v4f a2 = *(const v4f*)(arow + kb + 16);
            v4f a3 = *(const v4f*)(arow + kb + 20);
            v4f b00 = *(const v4f*)(brow0 + kb);
            v4f b01 = *(const v4f*)(brow0 + kb + 4);
            v4f b02 = *(const v4f*)(brow0 + kb + 16);
            v4f b03 = *(const v4f*)(brow0 + kb + 20);
            v4f b10 = *(const v4f*)(brow1 + kb);
            v4f b11 = *(const v4f*)(brow1 + kb + 4);
            v4f b12 = *(const v4f*)(brow1 + kb + 16);
            v4f b13 = *(const v4f*)(brow1 + kb + 20);
            v16bf a, bv0, bv1;
            #pragma unroll
            for (int i = 0; i < 4; ++i) {
                a[i]      = (__bf16)a0[i];
                a[4 + i]  = (__bf16)a1[i];
                a[8 + i]  = (__bf16)a2[i];
                a[12 + i] = (__bf16)a3[i];
                bv0[i]      = (__bf16)b00[i];
                bv0[4 + i]  = (__bf16)b01[i];
                bv0[8 + i]  = (__bf16)b02[i];
                bv0[12 + i] = (__bf16)b03[i];
                bv1[i]      = (__bf16)b10[i];
                bv1[4 + i]  = (__bf16)b11[i];
                bv1[8 + i]  = (__bf16)b12[i];
                bv1[12 + i] = (__bf16)b13[i];
            }
            c0 = __builtin_amdgcn_wmma_f32_16x16x32_bf16(
                     false, a, false, bv0, (short)0, c0, false, false);
            c1 = __builtin_amdgcn_wmma_f32_16x16x32_bf16(
                     false, a, false, bv1, (short)0, c1, false, false);
        }
    }

    // Epilogue. C/D layout: VGPR r, lanes 0-15: M=r; lanes 16-31: M=8+r; N=lane&15.
    const float bb0 = bias[col0];
    const float bb1 = bias[col1];

    #pragma unroll
    for (int r = 0; r < 8; ++r) {
        const int row = m_tile * 16 + r + sel * 8;
        const size_t o0 = (size_t)row * HID + col0;
        const size_t o1 = (size_t)row * HID + col1;
        h_out[o0] = fast_tanh(c0[r] + x_t[o0] + bb0);
        h_out[o1] = fast_tanh(c1[r] + x_t[o1] + bb1);
    }
}

extern "C" void kernel_launch(void* const* d_in, const int* in_sizes, int n_in,
                              void* d_out, int out_size, void* d_ws, size_t ws_size,
                              hipStream_t stream) {
    const float* input_proj = (const float*)d_in[0]; // (S, B, H)
    const float* hidden     = (const float*)d_in[1]; // (B, H)
    const float* weight_hh  = (const float*)d_in[2]; // (H, H)
    const float* bias_hh    = (const float*)d_in[3]; // (H)
    float* out = (float*)d_out;                      // (S, B, H)

    const size_t step    = (size_t)BATCH * HID;
    const size_t w_bytes = (size_t)HID * HID * sizeof(__bf16);
    const bool   preconv = (ws_size >= w_bytes);

    if (preconv) {
        convert_w_kernel<<<(HID * HID / 8) / 256, 256, 0, stream>>>(
            weight_hh, (__bf16*)d_ws);
    }

    const float* hprev = hidden;
    for (int t = 0; t < SEQ; ++t) {
        if (preconv) {
            rnn_step_kernel<true><<<128, 64, 0, stream>>>(
                input_proj + (size_t)t * step, hprev, d_ws, bias_hh,
                out + (size_t)t * step);
        } else {
            rnn_step_kernel<false><<<128, 64, 0, stream>>>(
                input_proj + (size_t)t * step, hprev, weight_hh, bias_hh,
                out + (size_t)t * step);
        }
        hprev = out + (size_t)t * step;  // kernel boundary = device-wide sync
    }
}
